// MDM_90331752169851
// MI455X (gfx1250) — compile-verified, matrix operand order checked by reading, not run on hardware
//
#include <hip/hip_runtime.h>

// Shapes
#define T_ 4096
#define B_ 32
#define D_ 512
#define H_ 8
#define HD_ 64
#define NROWS (B_ * T_)   // 131072

typedef __attribute__((ext_vector_type(16))) _Float16 v16h;
typedef __attribute__((ext_vector_type(8)))  _Float16 v8h;
typedef __attribute__((ext_vector_type(4)))  _Float16 v4h;
typedef __attribute__((ext_vector_type(8)))  float    v8f;
typedef __attribute__((ext_vector_type(4)))  float    v4f;

__device__ __forceinline__ v8f wmma16(v16h a, v16h b, v8f c) {
  // D = A(16x32 f16) x B(32x16 f16) + C(16x16 f32)
  return __builtin_amdgcn_wmma_f32_16x16x32_f16(false, a, false, b, (short)0, c,
                                                false, false);
}

// Load a 16-half fragment (A or B operand) from an LDS row.
// Per ISA layout: halves i<8 -> K = hi*8 + i ; i>=8 -> K = 16 + hi*8 + (i-8)
__device__ __forceinline__ v16h ldfrag(const _Float16* rowp, int hi) {
  v8h lo = *(const v8h*)(rowp + hi * 8);
  v8h up = *(const v8h*)(rowp + 16 + hi * 8);
  v16h r;
#pragma unroll
  for (int i = 0; i < 8; ++i) { r[i] = lo[i]; r[i + 8] = up[i]; }
  return r;
}

// Async DMA: 16 bytes per lane, global -> LDS, tracked by ASYNCcnt.
// Generic pointers to __shared__ carry the wave-relative LDS offset in the
// low 32 bits (ISA 10.2: only the low 32 address bits are used), so truncation is the
// correct VDST operand.
__device__ __forceinline__ void async_copy16(void* lds_dst, const void* gsrc) {
  unsigned d = (unsigned)(unsigned long long)lds_dst;
  asm volatile("global_load_async_to_lds_b128 %0, %1, off"
               :: "v"(d), "v"(gsrc) : "memory");
}
#define ASYNC_WAIT(n) asm volatile("s_wait_asynccnt " #n ::: "memory")

// ---------------- Kernel 0: one-time f32 -> f16 weight conversion -----------
__global__ __launch_bounds__(256) void k_wcvt(const float* __restrict__ Wq,
                                              const float* __restrict__ Wk,
                                              const float* __restrict__ Wv,
                                              _Float16* __restrict__ wh) {
  const float* src = (blockIdx.y == 0) ? Wq : (blockIdx.y == 1) ? Wk : Wv;
  size_t i = ((size_t)blockIdx.x * 256 + threadIdx.x) * 4;   // over 512*512
  v4f v = *(const v4f*)(src + i);
  v4h h;
#pragma unroll
  for (int c = 0; c < 4; ++c) h[c] = (_Float16)v[c];
  *(v4h*)(wh + (size_t)blockIdx.y * D_ * D_ + i) = h;
}

// ---------------- Kernel 1: per-row mean / rstd (shared by both LNs) --------
__global__ __launch_bounds__(256) void k_stats(const float* __restrict__ x,
                                               float2* __restrict__ stats) {
  int wave = threadIdx.x >> 5;
  int lane = threadIdx.x & 31;
  long R = (long)blockIdx.x * 8 + wave;          // flattened (b,t)
  int b = (int)(R >> 12);
  int t = (int)(R & (T_ - 1));
  const float* xr = x + ((size_t)t * B_ + b) * D_;
  float s = 0.f, ss = 0.f;
#pragma unroll
  for (int j = 0; j < 4; ++j) {
    v4f v = *(const v4f*)(xr + (lane + 32 * j) * 4);
#pragma unroll
    for (int c = 0; c < 4; ++c) { s += v[c]; ss += v[c] * v[c]; }
  }
#pragma unroll
  for (int off = 16; off; off >>= 1) {
    s  += __shfl_xor(s, off, 32);
    ss += __shfl_xor(ss, off, 32);
  }
  if (lane == 0) {
    float m = s * (1.0f / D_);
    float var = ss * (1.0f / D_) - m * m;
    stats[R] = make_float2(m, rsqrtf(var + 1e-5f));
  }
}

// ---------------- Kernel 2: fused LN + QKV projection (WMMA GEMM) ----------
// grid.x = NROWS/64 M-tiles, grid.y = 12 (0-3: Q | 4-7: K | 8-11: V)
// A tile: LN fused on VALU; B tile: f16 weights via async-to-LDS, dbl-buffered
__global__ __launch_bounds__(256) void k_qkv(
    const float* __restrict__ x, const float2* __restrict__ stats,
    const float* __restrict__ g1, const float* __restrict__ b1,
    const float* __restrict__ g2, const float* __restrict__ b2,
    const _Float16* __restrict__ wh,
    const float* __restrict__ bq, const float* __restrict__ bk,
    const float* __restrict__ bv,
    _Float16* __restrict__ qout, _Float16* __restrict__ kout,
    _Float16* __restrict__ vout) {
  __shared__ __align__(16) _Float16 At[64][40];       // [m][k]
  __shared__ __align__(16) _Float16 Bt[2][128][40];   // [buf][n][k]
  __shared__ float outb[64][136];                     // Q softmax staging

  int gy = blockIdx.y;
  int matsel = gy >> 2;                 // 0=Q 1=K 2=V
  int n0 = (gy & 3) * 128;
  const _Float16* W  = wh + (size_t)matsel * D_ * D_;
  const float* bias  = (matsel == 0) ? bq : (matsel == 1) ? bk : bv;
  const float* gg    = (matsel == 0) ? g1 : g2;
  const float* bb    = (matsel == 0) ? b1 : b2;

  long R0 = (long)blockIdx.x * 64;
  int tid = threadIdx.x;
  int lane = tid & 31, l15 = lane & 15, hi = (lane >> 4) & 1, w = tid >> 5;

  // this thread's two async B chunks: 128 n x 32 k halves = 512 x 16B
  int gB0 = tid, gB1 = tid + 256;
  int nB0 = gB0 >> 2, dgB0 = gB0 & 3;
  int nB1 = gB1 >> 2, dgB1 = gB1 & 3;

  v8f acc[4] = {};

  // prologue: kick off tile 0
  async_copy16(&Bt[0][nB0][dgB0 * 8], W + (size_t)(n0 + nB0) * D_ + dgB0 * 8);
  async_copy16(&Bt[0][nB1][dgB1 * 8], W + (size_t)(n0 + nB1) * D_ + dgB1 * 8);

  for (int kk = 0; kk < 16; ++kk) {
    int d0 = kk * 32;
    // Stage A: 64 rows x 32 d, LayerNorm fused, f32 -> f16 (VALU + ds_store)
#pragma unroll
    for (int j = 0; j < 2; ++j) {
      int g = tid + j * 256;            // 0..511
      int row = g >> 3, dg = g & 7;
      long R = R0 + row;
      int b = (int)(R >> 12), t = (int)(R & (T_ - 1));
      const float* xr = x + ((size_t)t * B_ + b) * D_ + d0 + dg * 4;
      if (kk + 1 < 16) __builtin_prefetch(xr + 32, 0, 0);   // global_prefetch_b8
      float2 st = stats[R];
      v4f xv = *(const v4f*)xr;
      v4h hv;
#pragma unroll
      for (int c = 0; c < 4; ++c) {
        int d = d0 + dg * 4 + c;
        hv[c] = (_Float16)((xv[c] - st.x) * st.y * gg[d] + bb[d]);
      }
      *(v4h*)&At[row][dg * 4] = hv;
    }
    // Prefetch next B tile into the other buffer (stays in flight over WMMA)
    if (kk + 1 < 16) {
      int buf = (kk + 1) & 1, dn = (kk + 1) * 32;
      async_copy16(&Bt[buf][nB0][dgB0 * 8],
                   W + (size_t)(n0 + nB0) * D_ + dn + dgB0 * 8);
      async_copy16(&Bt[buf][nB1][dgB1 * 8],
                   W + (size_t)(n0 + nB1) * D_ + dn + dgB1 * 8);
      ASYNC_WAIT(2);     // current tile's 2 loads done; next 2 still in flight
    } else {
      ASYNC_WAIT(0);
    }
    __syncthreads();

    const _Float16(*Bcur)[40] = Bt[kk & 1];
    v16h bf = ldfrag(&Bcur[w * 16 + l15][0], hi);
#pragma unroll
    for (int s = 0; s < 4; ++s) {
      v16h af = ldfrag(&At[s * 16 + l15][0], hi);
      acc[s] = wmma16(af, bf, acc[s]);
    }
    __syncthreads();
  }

  int nloc = w * 16 + l15;
  float bnv = bias[n0 + nloc];

  if (matsel == 0) {
    // Q: bias + softmax over the 64-wide head dim (tile = 2 full heads)
#pragma unroll
    for (int s = 0; s < 4; ++s)
#pragma unroll
      for (int r = 0; r < 8; ++r)
        outb[s * 16 + r + hi * 8][nloc] = acc[s][r] + bnv;
    __syncthreads();
    if (tid < 128) {
      int row = tid >> 1, head = tid & 1;
      float* p = &outb[row][head * 64];
      float mx = -1e30f;
      for (int j = 0; j < 64; ++j) mx = fmaxf(mx, p[j]);
      float sum = 0.f;
      for (int j = 0; j < 64; ++j) { float e = __expf(p[j] - mx); p[j] = e; sum += e; }
      float inv = 1.0f / sum;
      long R = R0 + row;
      int hg = (n0 >> 6) + head;
      _Float16* qp = qout + (size_t)R * D_ + hg * 64;
      for (int j = 0; j < 64; ++j) qp[j] = (_Float16)(p[j] * inv);
    }
  } else {
    _Float16* o = (matsel == 1) ? kout : vout;
#pragma unroll
    for (int s = 0; s < 4; ++s)
#pragma unroll
      for (int r = 0; r < 8; ++r) {
        long R = R0 + s * 16 + r + hi * 8;
        o[(size_t)R * D_ + n0 + nloc] = (_Float16)(acc[s][r] + bnv);
      }
  }
}

// ---------------- Kernel 3a/3b: column max of k over T ----------------------
__global__ __launch_bounds__(256) void k_cmax_part(const _Float16* __restrict__ k,
                                                   float* __restrict__ pmax) {
  int b = blockIdx.x, d = blockIdx.y * 256 + threadIdx.x, ch = blockIdx.z;
  const _Float16* kp = k + ((size_t)b * T_ + ch * 256) * D_ + d;
  float mx = -1e30f;
  for (int t = 0; t < 256; ++t) mx = fmaxf(mx, (float)kp[(size_t)t * D_]);
  pmax[((size_t)b * 16 + ch) * D_ + d] = mx;
}
__global__ __launch_bounds__(256) void k_cmax_fin(const float* __restrict__ pmax,
                                                  float* __restrict__ cmax) {
  int b = blockIdx.x, d = blockIdx.y * 256 + threadIdx.x;
  float mx = -1e30f;
  for (int ch = 0; ch < 16; ++ch)
    mx = fmaxf(mx, pmax[((size_t)b * 16 + ch) * D_ + d]);
  cmax[(size_t)b * D_ + d] = mx;
}

// ---------------- Kernel 4: state = (exp(k-max)^T V) / denom  ---------------
// one block per (b,h); state is 64x64, K-contraction over T via WMMA
__global__ __launch_bounds__(128) void k_attn(const _Float16* __restrict__ kws,
                                              const _Float16* __restrict__ vws,
                                              const float* __restrict__ cmax,
                                              _Float16* __restrict__ state) {
  __shared__ __align__(16) _Float16 ech[64][40];   // [d][t]  exp(k - max)
  __shared__ __align__(16) _Float16 vch[64][40];   // [l][t]
  __shared__ float denom[64];
  int bh = blockIdx.x, b = bh >> 3, h = bh & 7;
  int tid = threadIdx.x;
  int lane = tid & 31, l15 = lane & 15, hi = (lane >> 4) & 1, w = tid >> 5;
  if (tid < 64) denom[tid] = 0.f;
  int dl = tid & 63;                       // this thread's staged feature
  float cm = cmax[(size_t)b * D_ + h * 64 + dl];
  float dpart = 0.f;
  v8f acc[4] = {};
  size_t base = (size_t)b * T_ * D_ + h * 64;

  for (int c = 0; c < 128; ++c) {
    int t0 = c * 32;
#pragma unroll
    for (int j = 0; j < 16; ++j) {
      int t = (tid >> 6) + 2 * j;          // 0..31
      size_t off = base + (size_t)(t0 + t) * D_ + dl;
      float e = __expf((float)kws[off] - cm);
      dpart += e;
      ech[dl][t] = (_Float16)e;
      vch[dl][t] = vws[off];
    }
    __syncthreads();
    v16h af = ldfrag(&ech[w * 16 + l15][0], hi);
#pragma unroll
    for (int ns = 0; ns < 4; ++ns) {
      v16h bf = ldfrag(&vch[ns * 16 + l15][0], hi);
      acc[ns] = wmma16(af, bf, acc[ns]);
    }
    __syncthreads();
  }
  atomicAdd(&denom[dl], dpart);
  __syncthreads();
#pragma unroll
  for (int ns = 0; ns < 4; ++ns)
#pragma unroll
    for (int r = 0; r < 8; ++r) {
      int d = w * 16 + r + hi * 8;
      int l = ns * 16 + l15;
      state[((size_t)bh * 64 + l) * 64 + d] = (_Float16)(acc[ns][r] / denom[d]);
    }
}

// ---------------- Kernel 5: y = Q @ state (async-staged operands) -----------
__global__ __launch_bounds__(128) void k_y(const _Float16* __restrict__ qws,
                                           const _Float16* __restrict__ state,
                                           float* __restrict__ y) {
  __shared__ __align__(16) _Float16 Qt[64][72];   // [t][d]
  __shared__ __align__(16) _Float16 St[64][72];   // [l][d]
  int t0 = blockIdx.x * 64, bh = blockIdx.y, b = bh >> 3, h = bh & 7;
  int tid = threadIdx.x;
  int lane = tid & 31, l15 = lane & 15, hi = (lane >> 4) & 1, w = tid >> 5;
  size_t qbase = ((size_t)b * T_ + t0) * D_ + h * 64;
#pragma unroll
  for (int j = 0; j < 4; ++j) {
    int g = tid + 128 * j;                 // 0..511
    int row = g >> 3, dg = g & 7;
    async_copy16(&Qt[row][dg * 8], qws + qbase + (size_t)row * D_ + dg * 8);
    async_copy16(&St[row][dg * 8],
                 state + ((size_t)bh * 64 + row) * 64 + dg * 8);
  }
  ASYNC_WAIT(0);
  __syncthreads();
  v8f acc[4] = {};
#pragma unroll
  for (int kk = 0; kk < 2; ++kk) {
    v16h af = ldfrag(&Qt[w * 16 + l15][kk * 32], hi);
#pragma unroll
    for (int ns = 0; ns < 4; ++ns) {
      v16h bf = ldfrag(&St[ns * 16 + l15][kk * 32], hi);
      acc[ns] = wmma16(af, bf, acc[ns]);
    }
  }
#pragma unroll
  for (int ns = 0; ns < 4; ++ns)
#pragma unroll
    for (int r = 0; r < 8; ++r) {
      int t = t0 + w * 16 + r + hi * 8;
      y[((size_t)b * T_ + t) * D_ + h * 64 + ns * 16 + l15] = acc[ns][r];
    }
}

// ---------------- Host launcher ---------------------------------------------
extern "C" void kernel_launch(void* const* d_in, const int* in_sizes, int n_in,
                              void* d_out, int out_size, void* d_ws, size_t ws_size,
                              hipStream_t stream) {
  const float* x   = (const float*)d_in[0];
  const float* g1  = (const float*)d_in[1];
  const float* b1  = (const float*)d_in[2];
  const float* g2  = (const float*)d_in[3];
  const float* b2  = (const float*)d_in[4];
  const float* Wq  = (const float*)d_in[5];
  const float* bq  = (const float*)d_in[6];
  const float* Wk  = (const float*)d_in[7];
  const float* bk  = (const float*)d_in[8];
  const float* Wv  = (const float*)d_in[9];
  const float* bv  = (const float*)d_in[10];
  float* y = (float*)d_out;

  char* ws = (char*)d_ws;
  const size_t HALF_MAT = (size_t)NROWS * D_ * sizeof(_Float16); // 128 MB
  float2*   stats = (float2*)ws;                                  // 1 MB
  _Float16* qb    = (_Float16*)(ws + (1u << 20));
  _Float16* kb    = (_Float16*)((char*)qb + HALF_MAT);
  _Float16* vb    = (_Float16*)((char*)kb + HALF_MAT);
  float*    pmax  = (float*)((char*)vb + HALF_MAT);               // 1 MB
  float*    cmax  = pmax + (size_t)B_ * 16 * D_;                  // 64 KB
  _Float16* state = (_Float16*)(cmax + (size_t)B_ * D_);          // 2 MB
  _Float16* wh    = state + (size_t)B_ * H_ * HD_ * HD_;          // 1.5 MB

  k_wcvt<<<dim3(D_ * D_ / 1024, 3), dim3(256), 0, stream>>>(Wq, Wk, Wv, wh);

  k_stats<<<dim3(NROWS / 8), dim3(256), 0, stream>>>(x, stats);

  k_qkv<<<dim3(NROWS / 64, 12), dim3(256), 0, stream>>>(
      x, stats, g1, b1, g2, b2, wh, bq, bk, bv, qb, kb, vb);

  k_cmax_part<<<dim3(B_, 2, 16), dim3(256), 0, stream>>>(kb, pmax);
  k_cmax_fin<<<dim3(B_, 2), dim3(256), 0, stream>>>(pmax, cmax);

  k_attn<<<dim3(B_ * H_), dim3(128), 0, stream>>>(kb, vb, cmax, state);

  k_y<<<dim3(T_ / 64, B_ * H_), dim3(128), 0, stream>>>(qb, state, y);
}
